// Convpass_swintransformer_hypernet_mask_22359599743421
// MI455X (gfx1250) — compile-verified, matrix operand order checked by reading, not run on hardware
//
#include <hip/hip_runtime.h>
#include <hip/hip_bf16.h>
#include <math.h>

// ---- problem constants (match reference) ----
#define NB        64          // batch
#define HH        28
#define WWID      28
#define HW        784         // 28*28
#define C_DIM     384
#define DIM       64
#define META      64
#define NUM_MASK  16
#define KC        576         // DIM*9 (im2col K)

typedef __attribute__((ext_vector_type(16))) _Float16 v16h;
typedef __attribute__((ext_vector_type(8)))  _Float16 v8h;
typedef __attribute__((ext_vector_type(8)))  float    v8f;
typedef __attribute__((ext_vector_type(4)))  float    v4f;

__device__ __forceinline__ float qgelu_f(float v) {
  // x * sigmoid(1.702x); fast hw reciprocal (v_rcp_f32) instead of IEEE div chain
  const float e = __expf(-1.702f * v);
  return v * __builtin_amdgcn_rcpf(1.f + e);
}

// Fragment load: p -> element (row, k0+khalf); halves [p..p+7] and [p+16..p+23]
__device__ __forceinline__ v16h load_frag(const _Float16* p) {
  const v8h lo = *(const v8h*)(p);
  const v8h hi = *(const v8h*)(p + 16);
  v16h f;
#pragma unroll
  for (int e = 0; e < 8; ++e) { f[e] = lo[e]; f[e + 8] = hi[e]; }
  return f;
}

// ---------------------------------------------------------------------------
// WMMA GEMM on pre-converted f16 operands.
//   A  : f16 row-major [M,K]
//   Bt : f16 [Ncols>=ceil64(N)][K]  (B transposed; padded rows are zero)
//   C  : act(A@B + bias) -> f32 (OUT16=0) or f16 (OUT16=1)
// block = 128 (4 wave32), wave = 16x16 tile; grid (ceil(N/64), M/16).
// K multiple of 32. All load addresses are unconditionally in-bounds.
// ---------------------------------------------------------------------------
template <int ACT, int OUT16>
__global__ __launch_bounds__(128)
void gemm_f16_wmma(const _Float16* __restrict__ A, const _Float16* __restrict__ Bt,
                   const float* __restrict__ bias, float* __restrict__ C32,
                   _Float16* __restrict__ C16, int M, int N, int K) {
  const int lane  = threadIdx.x & 31;
  const int wave  = threadIdx.x >> 5;
  const int row0  = blockIdx.y * 16;
  const int col0  = blockIdx.x * 64 + wave * 16;
  const int khalf = (lane >> 4) << 3;
  const int arow  = row0 + (lane & 15);
  const int bcol  = col0 + (lane & 15);
  const _Float16* ap = A  + (size_t)arow * K + khalf;
  const _Float16* bp = Bt + (size_t)bcol * K + khalf;

  v8f acc = {};
  for (int k0 = 0; k0 < K; k0 += 32) {
    __builtin_prefetch((const void*)(ap + k0 + 128), 0, 0);
    const v16h af = load_frag(ap + k0);
    const v16h bf = load_frag(bp + k0);
    acc = __builtin_amdgcn_wmma_f32_16x16x32_f16(false, af, false, bf,
                                                 (short)0, acc, false, false);
  }

  if (bcol < N) {
    const float bv = bias ? bias[bcol] : 0.f;
    const int rbase = row0 + ((lane >> 4) << 3);
#pragma unroll
    for (int r = 0; r < 8; ++r) {
      float v = acc[r] + bv;
      if (ACT == 1) v = fmaxf(v, 0.f);
      if (ACT == 2) v = qgelu_f(v);
      if (OUT16) C16[(size_t)(rbase + r) * N + bcol] = (_Float16)v;
      else       C32[(size_t)(rbase + r) * N + bcol] = v;
    }
  }
}

// ---------------------------------------------------------------------------
// Per-sample 3x3 conv as im2col WMMA GEMM, A tile staged in LDS.
//   xact : f16 [B, HW, DIM]  (qgelu'd down-projection, channel-last)
//   cw   : f16 [B, DIM(out), KC]   (dynamic weights, already Bt layout)
//   yact : f16 [B, HW, DIM]  (qgelu applied)
// grid = (HW/16, B); block = 128 (4 waves x 16 out-channels).
// ---------------------------------------------------------------------------
__global__ __launch_bounds__(128)
void conv_gemm_wmma(const _Float16* __restrict__ xact,
                    const _Float16* __restrict__ cw,
                    _Float16* __restrict__ yact) {
  __shared__ _Float16 ldsA[16 * KC];              // 18 KB: 16 positions x 576
  const int b    = blockIdx.y;
  const int row0 = blockIdx.x * 16;
  const int tid  = threadIdx.x;

  // ---- cooperative im2col fill: thread = (pos, channel-octet) ----
  {
    const _Float16* xb = xact + (size_t)b * HW * DIM;
    const int pos = tid >> 3;                     // 0..15
    const int sub = tid & 7;                      // 0..7 -> 8 channels each
    const int p  = row0 + pos;                    // 784 = 49*16, always valid
    const int hh = p / WWID;
    const int ww = p - hh * WWID;
    _Float16* dst = ldsA + pos * KC + (sub * 8) * 9;   // 8 channels * 9 taps
#pragma unroll
    for (int t = 0; t < 9; ++t) {                 // taps: t/3,t%3 fold to consts
      const int nh = hh + t / 3 - 1;
      const int nw = ww + t % 3 - 1;
      if ((unsigned)nh < (unsigned)HH && (unsigned)nw < (unsigned)WWID) {
        const v8h v = *(const v8h*)(xb + (size_t)(nh * WWID + nw) * DIM + sub * 8);
#pragma unroll
        for (int j = 0; j < 8; ++j) dst[j * 9 + t] = v[j];
      } else {
#pragma unroll
        for (int j = 0; j < 8; ++j) dst[j * 9 + t] = (_Float16)0.f;
      }
    }
  }
  __syncthreads();

  // ---- WMMA over K=576 ----
  const int lane  = tid & 31;
  const int wave  = tid >> 5;
  const int khalf = (lane >> 4) << 3;
  const int n     = wave * 16 + (lane & 15);
  const _Float16* ap = ldsA + (lane & 15) * KC + khalf;
  const _Float16* bp = cw + (size_t)b * DIM * KC + (size_t)n * KC + khalf;

  v8f acc = {};
#pragma unroll 2
  for (int k0 = 0; k0 < KC; k0 += 32) {
    const v16h af = load_frag(ap + k0);           // ds_load_b128 x2
    const v16h bf = load_frag(bp + k0);           // global_load_b128 x2
    acc = __builtin_amdgcn_wmma_f32_16x16x32_f16(false, af, false, bf,
                                                 (short)0, acc, false, false);
  }

  const int rbase = row0 + ((lane >> 4) << 3);
#pragma unroll
  for (int r = 0; r < 8; ++r) {
    const float v = qgelu_f(acc[r]);
    yact[((size_t)b * HW + rbase + r) * DIM + n] = (_Float16)v;
  }
}

// ---------------------------------------------------------------------------
// Streaming f32 -> f16 convert (same layout). n multiple of 4.
// ---------------------------------------------------------------------------
__global__ __launch_bounds__(256)
void cvt_f16_kernel(const float* __restrict__ in, _Float16* __restrict__ out, int n4) {
  const int i = blockIdx.x * blockDim.x + threadIdx.x;
  if (i < n4) {
    const v4f v = *(const v4f*)(in + (size_t)i * 4);
    _Float16* o = out + (size_t)i * 4;
    o[0] = (_Float16)v.x; o[1] = (_Float16)v.y;
    o[2] = (_Float16)v.z; o[3] = (_Float16)v.w;
  }
}

// Transpose-convert: out[n*K + k] = (f16) in[k*N + n].  (B -> Bt)
__global__ __launch_bounds__(256)
void transpose_cvt_kernel(const float* __restrict__ in, _Float16* __restrict__ out,
                          int K, int N) {
  const int o = blockIdx.x * blockDim.x + threadIdx.x;
  if (o < K * N) {
    const int n = o / K, k = o - n * K;
    out[o] = (_Float16)in[(size_t)k * N + n];
  }
}

// mask_token [16,64] -> padded Bt [64][64] f16 (rows >=16 are zero).
// (Bt[n][k] = mask_token[n][k] since B = mask_token^T.)
__global__ void mask_pad_kernel(const float* __restrict__ mt, _Float16* __restrict__ mtP) {
  for (int o = threadIdx.x; o < 64 * META; o += blockDim.x) {
    const int n = o >> 6;
    mtP[o] = (n < NUM_MASK) ? (_Float16)mt[o] : (_Float16)0.f;
  }
}

// ---------------------------------------------------------------------------
// Softmax stats over spatial dim per (b, m): {max, 1/sum(exp)}.
// logits: f32 [B*HW, 16]. grid = (NUM_MASK, B), block = 256.
// ---------------------------------------------------------------------------
__global__ __launch_bounds__(256)
void softmax_stats(const float* __restrict__ logits, float* __restrict__ stats) {
  __shared__ float sred[256];
  const int m = blockIdx.x, b = blockIdx.y, t = threadIdx.x;
  const float* base = logits + (size_t)b * HW * NUM_MASK + m;

  float mx = -3.4e38f;
  for (int p = t; p < HW; p += 256) mx = fmaxf(mx, base[(size_t)p * NUM_MASK]);
  sred[t] = mx; __syncthreads();
  for (int s = 128; s > 0; s >>= 1) {
    if (t < s) sred[t] = fmaxf(sred[t], sred[t + s]);
    __syncthreads();
  }
  const float gmax = sred[0]; __syncthreads();

  float se = 0.f;
  for (int p = t; p < HW; p += 256) se += __expf(base[(size_t)p * NUM_MASK] - gmax);
  sred[t] = se; __syncthreads();
  for (int s = 128; s > 0; s >>= 1) {
    if (t < s) sred[t] += sred[t + s];
    __syncthreads();
  }
  if (t == 0) {
    stats[((size_t)b * NUM_MASK + m) * 2 + 0] = gmax;
    stats[((size_t)b * NUM_MASK + m) * 2 + 1] = __builtin_amdgcn_rcpf(sred[0]);
  }
}

// mask_sum[b,p] = sum_m softmax_m(logits)[p]. grid = B, block = 256.
__global__ __launch_bounds__(256)
void mask_sum_kernel(const float* __restrict__ logits, const float* __restrict__ stats,
                     float* __restrict__ msum) {
  const int b = blockIdx.x;
  float mx[NUM_MASK], inv[NUM_MASK];
#pragma unroll
  for (int m = 0; m < NUM_MASK; ++m) {
    mx[m]  = stats[((size_t)b * NUM_MASK + m) * 2 + 0];
    inv[m] = stats[((size_t)b * NUM_MASK + m) * 2 + 1];
  }
  for (int p = threadIdx.x; p < HW; p += blockDim.x) {
    const float* lr = logits + ((size_t)b * HW + p) * NUM_MASK;
    float s = 0.f;
#pragma unroll
    for (int m = 0; m < NUM_MASK; ++m) s += __expf(lr[m] - mx[m]) * inv[m];
    msum[(size_t)b * HW + p] = s;
  }
}

// feat[b,n] = sum_p msum[b,p] * prompt_f16[b,p,n] -> f16. grid=B, block=256.
__global__ __launch_bounds__(256)
void feat_kernel(const _Float16* __restrict__ prompt, const float* __restrict__ msum,
                 _Float16* __restrict__ feat) {
  __shared__ float s[256];
  const int b = blockIdx.x, t = threadIdx.x;
  const int n = t & 63, g = t >> 6;
  float acc = 0.f;
  for (int p = g; p < HW; p += 4)
    acc += msum[(size_t)b * HW + p] * (float)prompt[((size_t)b * HW + p) * META + n];
  s[t] = acc; __syncthreads();
  if (t < 64)
    feat[(size_t)b * META + t] = (_Float16)(s[t] + s[t + 64] + s[t + 128] + s[t + 192]);
}

// ---------------------------------------------------------------------------
extern "C" void kernel_launch(void* const* d_in, const int* in_sizes, int n_in,
                              void* d_out, int out_size, void* d_ws, size_t ws_size,
                              hipStream_t stream) {
  (void)in_sizes; (void)n_in; (void)out_size; (void)ws_size;

  const float* x          = (const float*)d_in[0];
  const float* meta_w1    = (const float*)d_in[1];
  const float* meta_b1    = (const float*)d_in[2];
  const float* meta_w2    = (const float*)d_in[3];
  const float* meta_b2    = (const float*)d_in[4];
  const float* mask_token = (const float*)d_in[5];
  const float* hyper_w    = (const float*)d_in[6];
  const float* hyper_b    = (const float*)d_in[7];
  const float* down_w     = (const float*)d_in[8];
  const float* down_b     = (const float*)d_in[9];
  const float* up_w       = (const float*)d_in[10];
  const float* up_b       = (const float*)d_in[11];
  float* out = (float*)d_out;

  const size_t ROWS = (size_t)NB * HW;  // 50176
  const int NHYP = DIM * DIM * 9;       // 36864

  // ---- workspace carve-up (all offsets 16B aligned) ----
  char* w = (char*)d_ws;
  float* buf_lg = (float*)w;          w += ROWS * NUM_MASK * 4;   // logits f32
  float* buf_st = (float*)w;          w += 2048 * 4;              // softmax stats
  float* buf_ms = (float*)w;          w += ROWS * 4;              // mask_sum
  _Float16* xf16  = (_Float16*)w;     w += ROWS * C_DIM * 2;      // x (f16)
  _Float16* w1T   = (_Float16*)w;     w += (size_t)META * C_DIM * 2;
  _Float16* w2T   = (_Float16*)w;     w += (size_t)META * META * 2;
  _Float16* mtP   = (_Float16*)w;     w += (size_t)64 * META * 2; // padded mask Bt
  _Float16* hypT  = (_Float16*)w;     w += (size_t)NHYP * META * 2;
  _Float16* dwT   = (_Float16*)w;     w += (size_t)DIM * C_DIM * 2;
  _Float16* uwT   = (_Float16*)w;     w += (size_t)C_DIM * DIM * 2;
  _Float16* hF16  = (_Float16*)w;     w += ROWS * META * 2;       // h -> reused x_down
  _Float16* prF16 = (_Float16*)w;     w += ROWS * META * 2;       // prompt -> reused y
  _Float16* ftF16 = (_Float16*)w;     w += (size_t)NB * META * 2; // feat
  _Float16* cwF16 = (_Float16*)w;     w += (size_t)NB * NHYP * 2; // dyn conv weights

  const dim3 blk(128), blk256(256);
  const dim3 gRows(1, (unsigned)(ROWS / 16));

  // ---- one-time f16 conversions ----
  {
    const int n4 = (int)(ROWS * C_DIM / 4);
    cvt_f16_kernel<<<dim3((n4 + 255) / 256), blk256, 0, stream>>>(x, xf16, n4);
  }
  transpose_cvt_kernel<<<dim3((C_DIM * META + 255) / 256), blk256, 0, stream>>>(
      meta_w1, w1T, C_DIM, META);
  transpose_cvt_kernel<<<dim3((META * META + 255) / 256), blk256, 0, stream>>>(
      meta_w2, w2T, META, META);
  mask_pad_kernel<<<dim3(1), blk256, 0, stream>>>(mask_token, mtP);
  transpose_cvt_kernel<<<dim3((META * NHYP + 255) / 256), blk256, 0, stream>>>(
      hyper_w, hypT, META, NHYP);
  transpose_cvt_kernel<<<dim3((C_DIM * DIM + 255) / 256), blk256, 0, stream>>>(
      down_w, dwT, C_DIM, DIM);
  transpose_cvt_kernel<<<dim3((DIM * C_DIM + 255) / 256), blk256, 0, stream>>>(
      up_w, uwT, DIM, C_DIM);

  // 1) h = relu(x @ meta_w1 + b1)           f16 out
  gemm_f16_wmma<1, 1><<<gRows, blk, 0, stream>>>(xf16, w1T, meta_b1, nullptr, hF16,
                                                 (int)ROWS, META, C_DIM);
  // 2) prompt = h @ meta_w2 + b2            f16 out
  gemm_f16_wmma<0, 1><<<gRows, blk, 0, stream>>>(hF16, w2T, meta_b2, nullptr, prF16,
                                                 (int)ROWS, META, META);
  // 3) logits = prompt @ mask_token^T       f32 out, N=16 (padded Bt loads)
  gemm_f16_wmma<0, 0><<<gRows, blk, 0, stream>>>(prF16, mtP, nullptr, buf_lg, nullptr,
                                                 (int)ROWS, NUM_MASK, META);
  // 4) per-(b,m) spatial softmax stats
  softmax_stats<<<dim3(NUM_MASK, NB), blk256, 0, stream>>>(buf_lg, buf_st);
  // 5) mask_sum + pooled feature (f16)
  mask_sum_kernel<<<dim3(NB), blk256, 0, stream>>>(buf_lg, buf_st, buf_ms);
  feat_kernel<<<dim3(NB), blk256, 0, stream>>>(prF16, buf_ms, ftF16);
  // 6) conv_w = feat @ hyper_w + hyper_b    f16 out == per-sample Bt layout [64][576]
  gemm_f16_wmma<0, 1><<<dim3(NHYP / 64, NB / 16), blk, 0, stream>>>(
      ftF16, hypT, hyper_b, nullptr, cwF16, NB, NHYP, META);
  // 7) x_down = qgelu(x @ down_w + b)       f16 out (reuse h buffer)
  gemm_f16_wmma<2, 1><<<gRows, blk, 0, stream>>>(xf16, dwT, down_b, nullptr, hF16,
                                                 (int)ROWS, DIM, C_DIM);
  // 8) per-sample 3x3 conv (LDS im2col) + qgelu -> f16 (reuse prompt buffer)
  conv_gemm_wmma<<<dim3(HW / 16, NB), blk, 0, stream>>>(hF16, cwF16, prF16);
  // 9) out = y @ up_w + up_b                f32 out
  gemm_f16_wmma<0, 0><<<dim3(C_DIM / 64, (unsigned)(ROWS / 16)), blk, 0, stream>>>(
      prF16, uwT, up_b, out, nullptr, (int)ROWS, C_DIM, DIM);
}
